// BertSelfAttention_25572235280919
// MI455X (gfx1250) — matrix operand order repaired and checked by reading, not compiled, not run on hardware
//
#include <hip/hip_runtime.h>

typedef __attribute__((ext_vector_type(16))) __bf16 bf16x16;
typedef __attribute__((ext_vector_type(8)))  float  f32x8;
typedef __attribute__((ext_vector_type(4)))  unsigned int u32x4;

union Frag {
  bf16x16 v;
  u32x4   q[2];
  __bf16  h[16];
};

__device__ __forceinline__ f32x8 wmma_bf16(const Frag& a, const Frag& b, f32x8 c) {
  // D = A(16x32 bf16) x B(32x16 bf16) + C(16x16 f32)
  return __builtin_amdgcn_wmma_f32_16x16x32_bf16(
      /*neg_a=*/false, a.v, /*neg_b=*/false, b.v,
      /*c_mod=*/(short)0, c, /*reuse_a=*/false, /*reuse_b=*/false);
}

// ---------------------------------------------------------------------------
// CDNA5 async memory->LDS copy (ASYNCcnt-tracked, no VGPR round trip).
// VDST VGPR = per-lane LDS byte address (low 32 bits of the generic pointer),
// VADDR = per-lane 64-bit global address.  [CDNA5 ISA]
// ---------------------------------------------------------------------------
__device__ __forceinline__ void async_b128(void* lds_dst, const void* gsrc) {
  const unsigned off = (unsigned)(unsigned long long)lds_dst;
  asm volatile("global_load_async_to_lds_b128 %0, %1, off"
               :
               : "v"(off), "v"(gsrc)
               : "memory");
}
__device__ __forceinline__ void wait_async0() {
  asm volatile("s_wait_asynccnt 0" ::: "memory");
}

// ---------------------------------------------------------------------------
// fp32 -> bf16 conversion (vectorized: float4 in, 4x bf16 (8B) out)
// ---------------------------------------------------------------------------
__global__ void f32_to_bf16_kernel(const float* __restrict__ src,
                                   __bf16* __restrict__ dst, int n4) {
  int i = blockIdx.x * blockDim.x + threadIdx.x;
  if (i < n4) {
    const float4 f = ((const float4*)src)[i];
    union { __bf16 h[4]; unsigned long long u; } o;
    o.h[0] = (__bf16)f.x; o.h[1] = (__bf16)f.y;
    o.h[2] = (__bf16)f.z; o.h[3] = (__bf16)f.w;
    ((unsigned long long*)dst)[i] = o.u;
  }
}

// ---------------------------------------------------------------------------
// GEMM: [4096,1024](bf16) x [1024,1024](bf16, row-major [k][n]) + bias
// MODE 0: write bf16 split-head layout [B*H, S, DH]
// MODE 1: write fp32 merged layout [4096, 1024]
// Block tile 128x64, BK=32, 256 threads = 8 waves, wave tile 32x32.
// Double-buffered LDS; X tile staged with async-to-LDS, W tile staged with a
// register transpose (async cannot transpose).
// ---------------------------------------------------------------------------
#define BM 128
#define BN 64
#define BK 32

template <int MODE>
__global__ __launch_bounds__(256)
void gemm_bf16_kernel(const __bf16* __restrict__ X,
                      const __bf16* __restrict__ W,
                      const float*  __restrict__ bias,
                      void* __restrict__ dst) {
  __shared__ __bf16 x_lds[2][BM][BK + 8];    // [m][k], row stride 80B
  __shared__ __bf16 wt_lds[2][BN][BK + 8];   // [n][k] (W transposed)

  const int tid  = threadIdx.x;
  const int lane = tid & 31;
  const int wid  = tid >> 5;
  const int lo   = lane & 15;
  const int hi   = lane >> 4;

  const int m0  = blockIdx.y * BM;
  const int n0  = blockIdx.x * BN;
  const int wm0 = (wid & 3) * 32;
  const int wn0 = (wid >> 2) * 32;

  // staging coordinates
  const int xr = tid >> 1, xc = (tid & 1) * 16;   // X: 2 thr/row, 32B each
  const int wk = tid >> 3, wn = (tid & 7) * 8;    // W: 8 thr/row of 32

  f32x8 zero = {};
  f32x8 acc[2][2];
#pragma unroll
  for (int i = 0; i < 2; ++i)
#pragma unroll
    for (int j = 0; j < 2; ++j) acc[i][j] = zero;

  auto stage = [&](int buf, int kc) {
    // X tile 128x32: pure copy -> async DMA to LDS
    const __bf16* xs = X + (size_t)(m0 + xr) * 1024 + kc + xc;
    async_b128(&x_lds[buf][xr][xc],     xs);
    async_b128(&x_lds[buf][xr][xc + 8], xs + 8);
    // W tile 32x64 transposed -> wt_lds[n][k] (register scatter)
    union { u32x4 q; __bf16 h[8]; } t;
    t.q = *(const u32x4*)(W + (size_t)(kc + wk) * 1024 + n0 + wn);
#pragma unroll
    for (int j = 0; j < 8; ++j) wt_lds[buf][wn + j][wk] = t.h[j];
  };

  stage(0, 0);
  wait_async0();
  __syncthreads();

  for (int it = 0; it < 1024 / BK; ++it) {
    const int cur = it & 1;
    if (it + 1 < 1024 / BK) stage(cur ^ 1, (it + 1) * BK);  // overlap DMA

    // A frags: row = lo (both halves), elems 0-7: K=kb0.., 8-15: K=16+kb0..
    Frag a[2], b[2];
    const int kb0 = hi * 8;
#pragma unroll
    for (int im = 0; im < 2; ++im) {
      const int r = wm0 + im * 16 + lo;
      a[im].q[0] = *(const u32x4*)&x_lds[cur][r][kb0];
      a[im].q[1] = *(const u32x4*)&x_lds[cur][r][16 + kb0];
    }
    // B frags: col = lo, elems 0-7: K=kbb.., 8-15: K=16+kbb..
    const int kbb = hi * 8;
#pragma unroll
    for (int in = 0; in < 2; ++in) {
      const int n = wn0 + in * 16 + lo;
      b[in].q[0] = *(const u32x4*)&wt_lds[cur][n][kbb];
      b[in].q[1] = *(const u32x4*)&wt_lds[cur][n][kbb + 16];
    }
#pragma unroll
    for (int im = 0; im < 2; ++im)
#pragma unroll
      for (int in = 0; in < 2; ++in)
        acc[im][in] = wmma_bf16(a[im], b[in], acc[im][in]);

    wait_async0();    // next buffer's DMA has landed
    __syncthreads();  // + all waves done with cur / done scattering W
  }

  // epilogue: C layout elem i -> row = 8*hi + i, col = lo
#pragma unroll
  for (int im = 0; im < 2; ++im) {
#pragma unroll
    for (int in = 0; in < 2; ++in) {
      const int gn = n0 + wn0 + in * 16 + lo;
      const float bv = bias[gn];
#pragma unroll
      for (int i = 0; i < 8; ++i) {
        const int gm = m0 + wm0 + im * 16 + hi * 8 + i;
        const float val = acc[im][in][i] + bv;
        if (MODE == 0) {
          const int bb = gm >> 11, s  = gm & 2047;
          const int h  = gn >> 6,  dh = gn & 63;
          ((__bf16*)dst)[(((size_t)(bb * 16 + h)) * 2048 + s) * 64 + dh] =
              (__bf16)val;
        } else {
          ((float*)dst)[(size_t)gm * 1024 + gn] = val;
        }
      }
    }
  }
}

// ---------------------------------------------------------------------------
// Flash attention: grid (32 q-tiles, 32 heads), 128 threads = 4 waves.
// Each wave: 16 query rows, full DH=64 output accumulation, online softmax.
// K tile async-DMA'd to LDS; V tile register-transposed; double-buffered so
// the next tile's staging overlaps the current tile's 8 WMMAs + softmax.
// ---------------------------------------------------------------------------
__global__ __launch_bounds__(128)
void attention_kernel(const __bf16* __restrict__ Q,
                      const __bf16* __restrict__ K,
                      const __bf16* __restrict__ V,
                      const unsigned char* __restrict__ mask,  // [32][2048]
                      __bf16* __restrict__ out) {              // [4096][1024]
  __shared__ __bf16 k_lds[2][32][72];     // keys x dh (row stride 144B)
  __shared__ __bf16 vt_lds[2][64][40];    // dh x keys (V transposed)
  __shared__ __bf16 p_lds[4][16][40];     // per-wave P re-layout buffer

  const int tid  = threadIdx.x;
  const int lane = tid & 31;
  const int wid  = tid >> 5;
  const int lo   = lane & 15, hi = lane >> 4;

  const int hg  = blockIdx.y;                  // global head 0..31
  const int qr0 = blockIdx.x * 64 + wid * 16;  // query row base

  const __bf16* Qh = Q + (size_t)hg * 2048 * 64;
  const __bf16* Kh = K + (size_t)hg * 2048 * 64;
  const __bf16* Vh = V + (size_t)hg * 2048 * 64;
  const unsigned char* mh = mask + (size_t)hg * 2048;

  const int sr = tid >> 2, sc = (tid & 3) * 16;  // staging coords (32x64 tile)

  auto stage = [&](int buf, int kt) {
    // K tile: pure copy -> async DMA
    const __bf16* ksrc = Kh + (size_t)(kt + sr) * 64 + sc;
    async_b128(&k_lds[buf][sr][sc],     ksrc);
    async_b128(&k_lds[buf][sr][sc + 8], ksrc + 8);
    // V tile: transpose through registers
    union { u32x4 q[2]; __bf16 h[16]; } tv;
    const __bf16* vsrc = Vh + (size_t)(kt + sr) * 64 + sc;
    tv.q[0] = *(const u32x4*)(vsrc);
    tv.q[1] = *(const u32x4*)(vsrc + 8);
#pragma unroll
    for (int j = 0; j < 16; ++j) vt_lds[buf][sc + j][sr] = tv.h[j];
  };

  // Preload Q fragments, pre-scaled by 1/sqrt(64)=0.125 (exact in bf16)
  Frag qf[2];
#pragma unroll
  for (int ks = 0; ks < 2; ++ks) {
    const __bf16* src = Qh + (size_t)(qr0 + lo) * 64 + ks * 32 + hi * 8;
    qf[ks].q[0] = *(const u32x4*)(src);
    qf[ks].q[1] = *(const u32x4*)(src + 16);
#pragma unroll
    for (int e = 0; e < 16; ++e)
      qf[ks].h[e] = (__bf16)((float)qf[ks].h[e] * 0.125f);
  }

  f32x8 zero = {};
  f32x8 o_acc[4] = {zero, zero, zero, zero};
  float m_run[8], l_run[8];
#pragma unroll
  for (int i = 0; i < 8; ++i) { m_run[i] = -3.0e38f; l_run[i] = 0.f; }

  stage(0, 0);
  wait_async0();
  __syncthreads();

  for (int it = 0; it < 64; ++it) {
    const int cur = it & 1;
    const int kt  = it * 32;
    if (it + 1 < 64) stage(cur ^ 1, kt + 32);  // overlap next tile's staging

    // S[16x32] = Q @ K^T : 2 key sub-tiles x 2 k-steps
    f32x8 s_acc[2] = {zero, zero};
#pragma unroll
    for (int t = 0; t < 2; ++t) {
#pragma unroll
      for (int ks = 0; ks < 2; ++ks) {
        Frag kb;
        const int n = t * 16 + lo;
        kb.q[0] = *(const u32x4*)&k_lds[cur][n][ks * 32 + hi * 8];
        kb.q[1] = *(const u32x4*)&k_lds[cur][n][ks * 32 + hi * 8 + 16];
        s_acc[t] = wmma_bf16(qf[ks], kb, s_acc[t]);
      }
    }

    // attention mask (key-wise broadcast over queries)
#pragma unroll
    for (int t = 0; t < 2; ++t) {
      if (!mh[kt + t * 16 + lo]) {
#pragma unroll
        for (int i = 0; i < 8; ++i) s_acc[t][i] = -1e9f;
      }
    }

    // online softmax; row r = 8*hi + i lives across one 16-lane half-wave
#pragma unroll
    for (int i = 0; i < 8; ++i) {
      float rmax = fmaxf(s_acc[0][i], s_acc[1][i]);
#pragma unroll
      for (int off = 1; off < 16; off <<= 1)
        rmax = fmaxf(rmax, __shfl_xor(rmax, off, 16));
      const float m_new = fmaxf(m_run[i], rmax);
      const float alpha = __expf(m_run[i] - m_new);
      const float p0 = __expf(s_acc[0][i] - m_new);
      const float p1 = __expf(s_acc[1][i] - m_new);
      float rsum = p0 + p1;
#pragma unroll
      for (int off = 1; off < 16; off <<= 1)
        rsum += __shfl_xor(rsum, off, 16);
      l_run[i] = l_run[i] * alpha + rsum;
      m_run[i] = m_new;
#pragma unroll
      for (int c = 0; c < 4; ++c) o_acc[c][i] *= alpha;
      const int row = hi * 8 + i;
      p_lds[wid][row][lo]      = (__bf16)p0;
      p_lds[wid][row][16 + lo] = (__bf16)p1;
    }

    // re-layout P (C-layout -> A-layout) through per-wave LDS, then O += P @ V
    Frag pf;
    {
      const int kb0 = hi * 8;
      pf.q[0] = *(const u32x4*)&p_lds[wid][lo][kb0];
      pf.q[1] = *(const u32x4*)&p_lds[wid][lo][16 + kb0];
    }
#pragma unroll
    for (int c = 0; c < 4; ++c) {
      Frag vb;
      const int n = c * 16 + lo;
      vb.q[0] = *(const u32x4*)&vt_lds[cur][n][hi * 8];
      vb.q[1] = *(const u32x4*)&vt_lds[cur][n][hi * 8 + 16];
      o_acc[c] = wmma_bf16(pf, vb, o_acc[c]);
    }

    wait_async0();    // next K tile fully in LDS
    __syncthreads();  // all waves done reading cur, done V scatter
  }

  // finalize: divide by l, store merged-head layout [B, S, D] bf16
  const int bb = hg >> 4, h = hg & 15;
#pragma unroll
  for (int i = 0; i < 8; ++i) {
    const float inv = 1.0f / l_run[i];
    const int s = qr0 + hi * 8 + i;
    __bf16* dstp = out + ((size_t)(bb * 2048 + s)) * 1024 + h * 64;
#pragma unroll
    for (int c = 0; c < 4; ++c)
      dstp[c * 16 + lo] = (__bf16)(o_acc[c][i] * inv);
  }
}

// ---------------------------------------------------------------------------
// Host-side orchestration
// ---------------------------------------------------------------------------
extern "C" void kernel_launch(void* const* d_in, const int* in_sizes, int n_in,
                              void* d_out, int out_size, void* d_ws,
                              size_t ws_size, hipStream_t stream) {
  const float* hidden = (const float*)d_in[0];
  const unsigned char* mask = (const unsigned char*)d_in[1];  // jnp bool
  const float* W_q = (const float*)d_in[2];
  const float* b_q = (const float*)d_in[3];
  const float* W_k = (const float*)d_in[4];
  const float* b_k = (const float*)d_in[5];
  const float* W_v = (const float*)d_in[6];
  const float* b_v = (const float*)d_in[7];
  const float* W_o = (const float*)d_in[8];
  const float* b_o = (const float*)d_in[9];

  // Workspace layout (48 MB total, all MB-aligned)
  char* ws = (char*)d_ws;
  __bf16* Xbf = (__bf16*)(ws);                          // 8 MB  hidden bf16
  __bf16* Wbf = (__bf16*)(ws + ((size_t)8  << 20));     // 8 MB  4 weights bf16
  __bf16* Qbf = (__bf16*)(ws + ((size_t)16 << 20));     // 8 MB  [32,2048,64]
  __bf16* Kbf = (__bf16*)(ws + ((size_t)24 << 20));     // 8 MB
  __bf16* Vbf = (__bf16*)(ws + ((size_t)32 << 20));     // 8 MB
  __bf16* Abf = (__bf16*)(ws + ((size_t)40 << 20));     // 8 MB  merged attn

  const int NE = 4096 * 1024;
  const int WE = 1024 * 1024;

  // fp32 -> bf16 conversions (one-time; weights reused many times after)
  const int n4 = NE / 4, w4 = WE / 4;
  f32_to_bf16_kernel<<<(n4 + 255) / 256, 256, 0, stream>>>(hidden, Xbf, n4);
  f32_to_bf16_kernel<<<(w4 + 255) / 256, 256, 0, stream>>>(W_q, Wbf + 0 * WE, w4);
  f32_to_bf16_kernel<<<(w4 + 255) / 256, 256, 0, stream>>>(W_k, Wbf + 1 * WE, w4);
  f32_to_bf16_kernel<<<(w4 + 255) / 256, 256, 0, stream>>>(W_v, Wbf + 2 * WE, w4);
  f32_to_bf16_kernel<<<(w4 + 255) / 256, 256, 0, stream>>>(W_o, Wbf + 3 * WE, w4);

  const dim3 ggrid(1024 / BN, 4096 / BM);  // (16, 32)

  // QKV projections (bias fused, split-head bf16 output)
  gemm_bf16_kernel<0><<<ggrid, 256, 0, stream>>>(Xbf, Wbf + 0 * WE, b_q, Qbf);
  gemm_bf16_kernel<0><<<ggrid, 256, 0, stream>>>(Xbf, Wbf + 1 * WE, b_k, Kbf);
  gemm_bf16_kernel<0><<<ggrid, 256, 0, stream>>>(Xbf, Wbf + 2 * WE, b_v, Vbf);

  // Flash attention (scores never hit memory)
  attention_kernel<<<dim3(32, 32), 128, 0, stream>>>(Qbf, Kbf, Vbf, mask, Abf);

  // Output projection -> fp32 d_out
  gemm_bf16_kernel<1><<<ggrid, 256, 0, stream>>>(Abf, Wbf + 3 * WE, b_o,
                                                 (float*)d_out);
}
